// CPC_1984274891124
// MI455X (gfx1250) — matrix operand order repaired and checked by reading, not compiled, not run on hardware
//
#include <hip/hip_runtime.h>
#include <math.h>

#define N_ROWS 8192
#define C_CAT  16
#define NP     9216      // 144 tiles * 64 (worst-case padded row count)
#define NT_MAX 144

typedef __attribute__((ext_vector_type(16))) _Float16 v16h;
typedef __attribute__((ext_vector_type(8)))  _Float16 v8h;
typedef __attribute__((ext_vector_type(8)))  float    v8f;

#if __has_builtin(__builtin_amdgcn_s_wait_asynccnt)
#define WAIT_ASYNC() __builtin_amdgcn_s_wait_asynccnt(0)
#else
#define WAIT_ASYNC() asm volatile("s_wait_asynccnt 0" ::: "memory")
#endif

static __device__ __forceinline__ v8f wmma16(v16h a, v16h b, v8f c) {
  return __builtin_amdgcn_wmma_f32_16x16x32_f16(false, a, false, b, (short)0, c, false, false);
}

// One 16-byte async global->LDS copy per active lane (ASYNCcnt-tracked).
static __device__ __forceinline__ void async_b128(const _Float16* gsrc, _Float16* ldst) {
  unsigned int  loff = (unsigned int)(unsigned long long)ldst;  // addr[31:0] == LDS offset
  unsigned long long ga = (unsigned long long)gsrc;
  asm volatile("global_load_async_to_lds_b128 %0, %1, off"
               :: "v"(loff), "v"(ga) : "memory");
}

// Stage ROWS x 32 f16 tile (row-major, 32 halfs/row) global -> LDS asynchronously.
template<int ROWS>
static __device__ __forceinline__ void stage_async(_Float16* lds, const _Float16* src,
                                                   int srcStride, int row0, int k0) {
  constexpr int ITER = (ROWS * 4) / 256;  // 16B chunks / 256 threads
#pragma unroll
  for (int it = 0; it < ITER; ++it) {
    int idx = (int)threadIdx.x + it * 256;
    int r  = idx >> 2;
    int kc = (idx & 3) << 3;
    async_b128(src + (long)(row0 + r) * srcStride + k0 + kc, lds + r * 32 + kc);
  }
}

// A fragment: 16x32 f16 tile. ISA A-layout: lanes 0-15 hold K=0..7,16..23 of row
// (lane&15); lanes 16-31 hold K=8..15,24..31.
static __device__ __forceinline__ v16h frag_a(const _Float16* tile) {
  int lane = threadIdx.x & 31;
  const _Float16* q = tile + (lane & 15) * 32 + ((lane >> 4) << 3);
  v8h lo = *(const v8h*)q;
  v8h hi = *(const v8h*)(q + 16);
  return __builtin_shufflevector(lo, hi, 0,1,2,3,4,5,6,7,8,9,10,11,12,13,14,15);
}

// B fragment: 32x16 (KxN) from Bt tile stored [n][k]. ISA B-layout: lanes 0-15 hold
// K=0..7,16..23 of column (lane&15); lanes 16-31 hold K=8..15,24..31.
static __device__ __forceinline__ v16h frag_b(const _Float16* tile) {
  int lane = threadIdx.x & 31;
  const _Float16* q = tile + (lane & 15) * 32 + ((lane >> 4) << 3);
  v8h lo = *(const v8h*)q;
  v8h hi = *(const v8h*)(q + 16);
  return __builtin_shufflevector(lo, hi, 0,1,2,3,4,5,6,7,8,9,10,11,12,13,14,15);
}

static __device__ __forceinline__ float softplusf(float v) {
  if (v > 20.0f) return v;
  float e = expf(v);
  if (v < -20.0f) return e;
  return log1pf(e);
}

// Double-buffered 128x128-tile GEMM core: C(128x128) += A(128xK) * Bt(128xK)^T
template<int KTOT>
static __device__ __forceinline__ void core_128(const _Float16* A, int lda, int arow0,
                                                const _Float16* Bt, int ldb, int brow0,
                                                _Float16 (&lA)[2][4096],
                                                _Float16 (&lB)[2][4096],
                                                v8f (&acc)[2][4], int r0, int cb) {
  constexpr int NK = KTOT / 32;
  stage_async<128>(lA[0], A, lda, arow0, 0);
  stage_async<128>(lB[0], Bt, ldb, brow0, 0);
  WAIT_ASYNC();
  __syncthreads();
  for (int ks = 0; ks < NK; ++ks) {
    int cur = ks & 1, nxt = cur ^ 1;
    if (ks + 1 < NK) {
      stage_async<128>(lA[nxt], A, lda, arow0, (ks + 1) * 32);
      stage_async<128>(lB[nxt], Bt, ldb, brow0, (ks + 1) * 32);
    }
    v16h a0 = frag_a(lA[cur] + r0 * 32);
    v16h a1 = frag_a(lA[cur] + (r0 + 16) * 32);
#pragma unroll
    for (int j = 0; j < 4; ++j) {
      v16h b = frag_b(lB[cur] + (cb + j * 16) * 32);
      acc[0][j] = wmma16(a0, b, acc[0][j]);
      acc[1][j] = wmma16(a1, b, acc[1][j]);
    }
    WAIT_ASYNC();
    __syncthreads();
  }
}

// ---------------------------------------------------------------------------
// meta (ints): [0..15] cnt, [16..31] base (64-aligned sorted offset),
// [32..47] ntiles per cat, [48] NT_total, [64..207] tile -> category
// ---------------------------------------------------------------------------
__global__ __launch_bounds__(256) void k_setup(const int* __restrict__ c,
                                               int* __restrict__ meta,
                                               int* __restrict__ pos) {
  __shared__ int scnt[C_CAT], sbase[C_CAT], soff[C_CAT];
  int t = threadIdx.x;
  if (t < C_CAT) scnt[t] = 0;
  __syncthreads();
  for (int n = t; n < N_ROWS; n += 256) atomicAdd(&scnt[c[n]], 1);
  __syncthreads();
  if (t == 0) {
    int acc = 0, tile = 0;
    for (int i = 0; i < C_CAT; ++i) {
      sbase[i] = acc;
      int nt = (scnt[i] + 63) >> 6;
      meta[32 + i] = nt;
      for (int k = 0; k < nt; ++k) meta[64 + tile++] = i;
      acc += nt << 6;
    }
    meta[48] = tile;
  }
  __syncthreads();
  if (t < C_CAT) {
    meta[t]      = scnt[t];
    meta[16 + t] = sbase[t];
    soff[t]      = sbase[t];
  }
  __syncthreads();
  for (int n = t; n < N_ROWS; n += 256) pos[n] = atomicAdd(&soff[c[n]], 1);
}

// Convert x,z to f16; transpose+convert weights to Bt f16; zero accumulators.
__global__ __launch_bounds__(256) void k_prep(const float* __restrict__ x,
                                              const float* __restrict__ z,
                                              const float* __restrict__ W1,
                                              const float* __restrict__ W2,
                                              const float* __restrict__ Wz,
                                              const float* __restrict__ ws,
                                              _Float16* __restrict__ x16,
                                              _Float16* __restrict__ z16,
                                              _Float16* __restrict__ W1t,
                                              _Float16* __restrict__ W2t,
                                              _Float16* __restrict__ Wzt,
                                              _Float16* __restrict__ wst,
                                              _Float16* __restrict__ fxp,
                                              _Float16* __restrict__ fzp,
                                              float* __restrict__ diag,
                                              float* __restrict__ negsum) {
  const long Sx = 8192L * 256, Sz = 8192L * 128;
  const long S1 = 512L * 256, S2 = 128L * 512, S3 = 128L * 128, S4 = 16L * 128 * 128;
  const long S5 = (long)NP * 128, S6 = NP;
  long i = (long)blockIdx.x * 256 + threadIdx.x;
  if (i < Sx) { x16[i] = (_Float16)x[i]; return; }
  i -= Sx;
  if (i < Sz) { z16[i] = (_Float16)z[i]; return; }
  i -= Sz;
  if (i < S1) { long n = i >> 8, k = i & 255;  W1t[i] = (_Float16)W1[k * 512 + n]; return; }
  i -= S1;
  if (i < S2) { long n = i >> 9, k = i & 511;  W2t[i] = (_Float16)W2[k * 128 + n]; return; }
  i -= S2;
  if (i < S3) { long n = i >> 7, k = i & 127;  Wzt[i] = (_Float16)Wz[k * 128 + n]; return; }
  i -= S3;
  if (i < S4) {
    long ct = i >> 14, r = i & 16383, e = r >> 7, d = r & 127;
    wst[i] = (_Float16)ws[(ct << 14) + d * 128 + e];
    return;
  }
  i -= S4;
  if (i < S5) { fxp[i] = (_Float16)0.0f; return; }
  i -= S5;
  if (i < S5) { fzp[i] = (_Float16)0.0f; return; }
  i -= S5;
  if (i < S6) { diag[i] = 0.0f; return; }
  i -= S6;
  if (i < S6) { negsum[i] = 0.0f; return; }
}

// H = relu(x @ W1 + b1).  grid (64,4), 256 thr, tile 128x128, K=256.
__global__ __launch_bounds__(256) void k_gemm1(const _Float16* __restrict__ x16,
                                               const _Float16* __restrict__ W1t,
                                               const float* __restrict__ b1,
                                               _Float16* __restrict__ H) {
  __shared__ _Float16 lA[2][4096];
  __shared__ _Float16 lB[2][4096];
  int mtile = blockIdx.x, ntile = blockIdx.y;
  int w = threadIdx.x >> 5, lane = threadIdx.x & 31;
  int r0 = (w & 3) * 32, cb = (w >> 2) * 64;
  v8f acc[2][4] = {};
  core_128<256>(x16, 256, mtile * 128, W1t, 256, ntile * 128, lA, lB, acc, r0, cb);
  int colB = ntile * 128 + cb + (lane & 15);
  int rowB = mtile * 128 + r0 + ((lane >> 4) << 3);
#pragma unroll
  for (int i = 0; i < 2; ++i)
#pragma unroll
    for (int j = 0; j < 4; ++j) {
      int col = colB + j * 16;
      float bias = b1[col];
#pragma unroll
      for (int v = 0; v < 8; ++v) {
        float val = acc[i][j][v] + bias;
        val = val > 0.0f ? val : 0.0f;
        H[(long)(rowB + i * 16 + v) * 512 + col] = (_Float16)val;
      }
    }
}

// f_x = H @ W2 + b2, scattered to sorted order. grid 64, K=512.
__global__ __launch_bounds__(256) void k_gemm2(const _Float16* __restrict__ H,
                                               const _Float16* __restrict__ W2t,
                                               const float* __restrict__ b2,
                                               const int* __restrict__ pos,
                                               _Float16* __restrict__ fxp) {
  __shared__ _Float16 lA[2][4096];
  __shared__ _Float16 lB[2][4096];
  int mtile = blockIdx.x;
  int w = threadIdx.x >> 5, lane = threadIdx.x & 31;
  int r0 = (w & 3) * 32, cb = (w >> 2) * 64;
  v8f acc[2][4] = {};
  core_128<512>(H, 512, mtile * 128, W2t, 512, 0, lA, lB, acc, r0, cb);
  int colB = cb + (lane & 15);
  int rowB = mtile * 128 + r0 + ((lane >> 4) << 3);
#pragma unroll
  for (int i = 0; i < 2; ++i)
#pragma unroll
    for (int j = 0; j < 4; ++j) {
      int col = colB + j * 16;
      float bias = b2[col];
#pragma unroll
      for (int v = 0; v < 8; ++v) {
        int drow = pos[rowB + i * 16 + v];
        fxp[(long)drow * 128 + col] = (_Float16)(acc[i][j][v] + bias);
      }
    }
}

// f_z = z @ Wz + bz, scattered to sorted order. grid 64, K=128.
__global__ __launch_bounds__(256) void k_gemm3(const _Float16* __restrict__ z16,
                                               const _Float16* __restrict__ Wzt,
                                               const float* __restrict__ bz,
                                               const int* __restrict__ pos,
                                               _Float16* __restrict__ fzp) {
  __shared__ _Float16 lA[2][4096];
  __shared__ _Float16 lB[2][4096];
  int mtile = blockIdx.x;
  int w = threadIdx.x >> 5, lane = threadIdx.x & 31;
  int r0 = (w & 3) * 32, cb = (w >> 2) * 64;
  v8f acc[2][4] = {};
  core_128<128>(z16, 128, mtile * 128, Wzt, 128, 0, lA, lB, acc, r0, cb);
  int colB = cb + (lane & 15);
  int rowB = mtile * 128 + r0 + ((lane >> 4) << 3);
#pragma unroll
  for (int i = 0; i < 2; ++i)
#pragma unroll
    for (int j = 0; j < 4; ++j) {
      int col = colB + j * 16;
      float bias = bz[col];
#pragma unroll
      for (int v = 0; v < 8; ++v) {
        int drow = pos[rowB + i * 16 + v];
        fzp[(long)drow * 128 + col] = (_Float16)(acc[i][j][v] + bias);
      }
    }
}

// u = fxp @ w_s[cat(tile)], per 64-row tile. grid NT_MAX, tile 64x128, K=128.
__global__ __launch_bounds__(256) void k_gemm_u(const _Float16* __restrict__ fxp,
                                                const _Float16* __restrict__ wst,
                                                const int* __restrict__ meta,
                                                _Float16* __restrict__ u) {
  __shared__ _Float16 lA[2][2048];
  __shared__ _Float16 lB[2][4096];
  int t = blockIdx.x;
  if (t >= meta[48]) return;
  int cat = meta[64 + t];
  const _Float16* B = wst + (long)cat * 16384;
  int w = threadIdx.x >> 5, lane = threadIdx.x & 31;
  int rf = (w & 3) * 16, cb = (w >> 2) * 64;
  v8f acc[4] = {};
  stage_async<64>(lA[0], fxp, 128, t * 64, 0);
  stage_async<128>(lB[0], B, 128, 0, 0);
  WAIT_ASYNC();
  __syncthreads();
  for (int ks = 0; ks < 4; ++ks) {
    int cur = ks & 1, nxt = cur ^ 1;
    if (ks < 3) {
      stage_async<64>(lA[nxt], fxp, 128, t * 64, (ks + 1) * 32);
      stage_async<128>(lB[nxt], B, 128, 0, (ks + 1) * 32);
    }
    v16h a = frag_a(lA[cur] + rf * 32);
#pragma unroll
    for (int j = 0; j < 4; ++j) {
      v16h b = frag_b(lB[cur] + (cb + j * 16) * 32);
      acc[j] = wmma16(a, b, acc[j]);
    }
    WAIT_ASYNC();
    __syncthreads();
  }
  int colB = cb + (lane & 15);
  int rowB = t * 64 + rf + ((lane >> 4) << 3);
#pragma unroll
  for (int j = 0; j < 4; ++j) {
    int col = colB + j * 16;
#pragma unroll
    for (int v = 0; v < 8; ++v)
      u[(long)(rowB + v) * 128 + col] = (_Float16)acc[j][v];
  }
}

// Block-diagonal scores S = u @ fzp^T within a category; fused softplus row
// reduction (negative term) + diagonal capture (positive term).
// grid (NT_MAX, 128), tile 64x64, K=128.
__global__ __launch_bounds__(256) void k_scores(const _Float16* __restrict__ u,
                                                const _Float16* __restrict__ fzp,
                                                const int* __restrict__ meta,
                                                float* __restrict__ diag,
                                                float* __restrict__ negsum) {
  __shared__ _Float16 lA[2][2048];
  __shared__ _Float16 lB[2][2048];
  __shared__ float rowsum[64];
  int ti = blockIdx.x, tj = blockIdx.y;
  if (ti >= meta[48]) return;
  int cat = meta[64 + ti];
  if (tj >= meta[32 + cat]) return;
  int cnt  = meta[cat];
  int base = meta[16 + cat];
  int gi0 = ti * 64;
  int cs0 = base + tj * 64;
  int w = threadIdx.x >> 5, lane = threadIdx.x & 31;
  int rf = (w & 3) * 16, ch = (w >> 2) * 32;
  if (threadIdx.x < 64) rowsum[threadIdx.x] = 0.0f;
  v8f acc[2] = {};
  stage_async<64>(lA[0], u, 128, gi0, 0);
  stage_async<64>(lB[0], fzp, 128, cs0, 0);
  WAIT_ASYNC();
  __syncthreads();
  for (int ks = 0; ks < 4; ++ks) {
    int cur = ks & 1, nxt = cur ^ 1;
    if (ks < 3) {
      stage_async<64>(lA[nxt], u, 128, gi0, (ks + 1) * 32);
      stage_async<64>(lB[nxt], fzp, 128, cs0, (ks + 1) * 32);
    }
    v16h a = frag_a(lA[cur] + rf * 32);
#pragma unroll
    for (int j = 0; j < 2; ++j) {
      v16h b = frag_b(lB[cur] + (ch + j * 16) * 32);
      acc[j] = wmma16(a, b, acc[j]);
    }
    WAIT_ASYNC();
    __syncthreads();
  }
  int rbase = rf + ((lane >> 4) << 3);
#pragma unroll
  for (int v = 0; v < 8; ++v) {
    int rowTile  = rbase + v;
    int gi       = gi0 + rowTile;
    int rowLocal = gi - base;
    if (rowLocal >= cnt) continue;
    float s = 0.0f;
#pragma unroll
    for (int j = 0; j < 2; ++j) {
      int colTile   = ch + j * 16 + (lane & 15);
      int gcolLocal = tj * 64 + colTile;
      if (gcolLocal < cnt) {
        float m = acc[j][v];
        s += softplusf(m);
        if (rowLocal == gcolLocal) diag[gi] = m;
      }
    }
    atomicAdd(&rowsum[rowTile], s);
  }
  __syncthreads();
  if (threadIdx.x < 64) atomicAdd(&negsum[gi0 + threadIdx.x], rowsum[threadIdx.x]);
}

__global__ __launch_bounds__(256) void k_final(const int* __restrict__ c,
                                               const int* __restrict__ pos,
                                               const int* __restrict__ meta,
                                               const float* __restrict__ diag,
                                               const float* __restrict__ negsum,
                                               float* __restrict__ out) {
  int n = blockIdx.x * 256 + threadIdx.x;
  if (n >= N_ROWS) return;
  int p  = pos[n];
  int cc = c[n];
  float T    = softplusf(diag[p]);
  float negT = negsum[p] / (float)meta[cc];
  out[n] = logf(T + 1e-16f) - logf(negT + 1e-16f);
}

extern "C" void kernel_launch(void* const* d_in, const int* in_sizes, int n_in,
                              void* d_out, int out_size, void* d_ws, size_t ws_size,
                              hipStream_t stream) {
  const float* x  = (const float*)d_in[0];
  const int*   c  = (const int*)d_in[1];
  const float* z  = (const float*)d_in[2];
  const float* W1 = (const float*)d_in[3];
  const float* b1 = (const float*)d_in[4];
  const float* W2 = (const float*)d_in[5];
  const float* b2 = (const float*)d_in[6];
  const float* Wz = (const float*)d_in[7];
  const float* bz = (const float*)d_in[8];
  const float* ws = (const float*)d_in[9];
  float* out = (float*)d_out;

  char* p = (char*)d_ws;
  size_t off = 0;
  _Float16* H    = (_Float16*)(p + off); off += (size_t)8192 * 512 * 2;
  _Float16* x16  = (_Float16*)(p + off); off += (size_t)8192 * 256 * 2;
  _Float16* z16  = (_Float16*)(p + off); off += (size_t)8192 * 128 * 2;
  _Float16* fxp  = (_Float16*)(p + off); off += (size_t)NP * 128 * 2;
  _Float16* fzp  = (_Float16*)(p + off); off += (size_t)NP * 128 * 2;
  _Float16* ubuf = (_Float16*)(p + off); off += (size_t)NP * 128 * 2;
  _Float16* W1t  = (_Float16*)(p + off); off += (size_t)512 * 256 * 2;
  _Float16* W2t  = (_Float16*)(p + off); off += (size_t)128 * 512 * 2;
  _Float16* Wzt  = (_Float16*)(p + off); off += (size_t)128 * 128 * 2;
  _Float16* wst  = (_Float16*)(p + off); off += (size_t)16 * 128 * 128 * 2;
  int*      pos  = (int*)(p + off);      off += (size_t)8192 * 4;
  int*      meta = (int*)(p + off);      off += 1024;
  float*    diag = (float*)(p + off);    off += (size_t)NP * 4;
  float*    nsum = (float*)(p + off);    off += (size_t)NP * 4;

  k_setup<<<1, 256, 0, stream>>>(c, meta, pos);

  const long prep_total = 8192L * 256 + 8192L * 128
                        + 512L * 256 + 128L * 512 + 128L * 128 + 16L * 128 * 128
                        + 2L * NP * 128 + 2L * NP;
  int prep_blocks = (int)((prep_total + 255) / 256);
  k_prep<<<prep_blocks, 256, 0, stream>>>(x, z, W1, W2, Wz, ws, x16, z16,
                                          W1t, W2t, Wzt, wst, fxp, fzp, diag, nsum);

  k_gemm1<<<dim3(64, 4), 256, 0, stream>>>(x16, W1t, b1, H);
  k_gemm2<<<64, 256, 0, stream>>>(H, W2t, b2, pos, fxp);
  k_gemm3<<<64, 256, 0, stream>>>(z16, Wzt, bz, pos, fzp);
  k_gemm_u<<<NT_MAX, 256, 0, stream>>>(fxp, wst, meta, ubuf);
  k_scores<<<dim3(NT_MAX, 128), 256, 0, stream>>>(ubuf, fzp, meta, diag, nsum);
  k_final<<<32, 256, 0, stream>>>(c, pos, meta, diag, nsum, out);
}
